// MolecularGatConv_44014824849806
// MI455X (gfx1250) — matrix operand order repaired and checked by reading, not compiled, not run on hardware
//
#include <hip/hip_runtime.h>
#include <hip/hip_bf16.h>
#include <math.h>

// ---------------------------------------------------------------------------
// MolecularGatConv for MI455X (gfx1250, wave32, WMMA)
// GEMMs run as bf16 WMMA (v_wmma_f32_16x16x32_bf16) with f32 accumulation.
// GEMM blocking: 64 rows/block, 8 waves each owning a 16-col strip; the
// per-wave B fragments (32 VGPRs) are loaded once and reused across 4 row
// tiles -> 16 WMMAs per wave, 4x less B traffic.
// ---------------------------------------------------------------------------

typedef __attribute__((ext_vector_type(16))) __bf16 v16bf;
typedef __attribute__((ext_vector_type(8)))  __bf16 v8bf;
typedef __attribute__((ext_vector_type(4)))  __bf16 v4bf;
typedef __attribute__((ext_vector_type(8)))  float  v8f;

#define HDIM 128
#define SLOPE 0.2f

__device__ __forceinline__ float wave_sum(float v) {
#pragma unroll
  for (int m = 16; m > 0; m >>= 1) v += __shfl_xor(v, m, 32);
  return v;
}

// Exact float atomic-max via integer ordering (deterministic).
__device__ __forceinline__ void atomic_max_float(float* addr, float val) {
  if (val >= 0.0f) atomicMax((int*)addr, __float_as_int(val));
  else             atomicMin((unsigned int*)addr, (unsigned int)__float_as_int(val));
}

// ---------------------------------------------------------------------------
// Small utility kernels
// ---------------------------------------------------------------------------
__global__ void __launch_bounds__(256) zero_f(float* __restrict__ p, int n) {
  for (int i = blockIdx.x * blockDim.x + threadIdx.x; i < n;
       i += gridDim.x * blockDim.x) p[i] = 0.0f;
}

__global__ void __launch_bounds__(256)
reduce_sum_k(const float* __restrict__ a, int n, float* __restrict__ dst) {
  float v = 0.0f;
  for (int i = blockIdx.x * blockDim.x + threadIdx.x; i < n;
       i += gridDim.x * blockDim.x) v += a[i];
  v = wave_sum(v);
  __shared__ float s[8];
  int lane = threadIdx.x & 31, wv = threadIdx.x >> 5;
  if (lane == 0) s[wv] = v;
  __syncthreads();
  if (wv == 0) {
    float t = (lane < (int)(blockDim.x >> 5)) ? s[lane] : 0.0f;
    t = wave_sum(t);
    if (lane == 0) atomicAdd(dst, t);
  }
}

// scal[0] = ea_mean, scal[1+l] = dot(edge_W[l], att_edge[l]), scal[4] = ea_sum
__global__ void prep_scalars_k(const float* __restrict__ edge_W,
                               const float* __restrict__ att_edge,
                               float* __restrict__ scal, int L, int E) {
  int wv = threadIdx.x >> 5, lane = threadIdx.x & 31;
  if (wv < L) {
    const float4 w4 = *(const float4*)(edge_W  + (size_t)wv * HDIM + lane * 4);
    const float4 a4 = *(const float4*)(att_edge + (size_t)wv * HDIM + lane * 4);
    float s = wave_sum(w4.x * a4.x + w4.y * a4.y + w4.z * a4.z + w4.w * a4.w);
    if (lane == 0) scal[1 + wv] = s;
  } else if (lane == 0) {
    scal[0] = scal[4] / (float)E;   // mean edge_attr (self-loop fill value)
  }
}

// Transposed bf16 weights: WbT[l][n*H+k] = W[l][k*H+n]  (gat_W then lin_W)
__global__ void __launch_bounds__(256)
convert_weights_k(const float* __restrict__ gat_W, const float* __restrict__ lin_W,
                  __bf16* __restrict__ WbT, int L, int LO) {
  int idx = blockIdx.x * blockDim.x + threadIdx.x;
  int tot = (L + LO) * HDIM * HDIM;
  if (idx >= tot) return;
  int l = idx / (HDIM * HDIM);
  int r = idx - l * HDIM * HDIM;
  int nn = r >> 7;
  int k  = r & 127;
  const float* Wsrc = (l < L) ? (gat_W + (size_t)l * HDIM * HDIM)
                              : (lin_W + (size_t)(l - L) * HDIM * HDIM);
  WbT[idx] = (__bf16)Wsrc[(size_t)k * HDIM + nn];
}

// h[n][:] = emb[x[n]][:]  (f32 + bf16 copies). One wave per node, 4 dims/lane.
__global__ void __launch_bounds__(256)
embed_k(const int* __restrict__ x, const float* __restrict__ emb,
        float* __restrict__ h, __bf16* __restrict__ hb, int N) {
  int t = blockIdx.x * blockDim.x + threadIdx.x;
  int n = t >> 5, lane = t & 31;
  if (n >= N) return;
  const float4 v = *(const float4*)(emb + (size_t)x[n] * HDIM + lane * 4);
  *(float4*)(h + (size_t)n * HDIM + lane * 4) = v;
  v4bf b; b[0] = (__bf16)v.x; b[1] = (__bf16)v.y; b[2] = (__bf16)v.z; b[3] = (__bf16)v.w;
  *(v4bf*)(hb + (size_t)n * HDIM + lane * 4) = b;
}

// ---------------------------------------------------------------------------
// bf16 WMMA GEMM: C[nrows x 128] = A[nrows x 128] * B[128 x 128]
// A row-major bf16, BT = B transposed ([n][k]) bf16. Optional bias+ReLU,
// optional bf16 re-emit. Block = 64 rows; wave = one 16-col strip; B frags
// held in registers across 4 row tiles (16 WMMAs / wave).
// ---------------------------------------------------------------------------
__global__ void __launch_bounds__(256)
gemm128_bf16_wmma(const __bf16* __restrict__ A, const __bf16* __restrict__ BT,
                  const float* __restrict__ bias, float* __restrict__ C,
                  __bf16* __restrict__ Cb, int nrows, int do_relu) {
  const int lane = threadIdx.x & 31;
  const int wave = threadIdx.x >> 5;
  const int rowBase = blockIdx.x * 64;
  const int colBase = wave * 16;
  const int hi  = lane >> 4;   // half-wave select
  const int l15 = lane & 15;

  // B 32x16 fragments, one per K-step, loaded once and reused for all row
  // tiles. Layout: lane = column, lanes 0-15 K kk..kk+15, 16-31 K kk+16..kk+31.
  const __bf16* bp = BT + (size_t)(colBase + l15) * HDIM;
  v16bf bfrag[4];
#pragma unroll
  for (int ks = 0; ks < 4; ++ks) {
    v8bf blo = *(const v8bf*)(bp + ks * 32 + hi * 16);
    v8bf bhi = *(const v8bf*)(bp + ks * 32 + hi * 16 + 8);
    bfrag[ks] = __builtin_shufflevector(blo, bhi,
                                        0,1,2,3,4,5,6,7,8,9,10,11,12,13,14,15);
  }
  const float bv = bias ? bias[colBase + l15] : 0.0f;

#pragma unroll
  for (int rt = 0; rt < 4; ++rt) {
    int tileRow = rowBase + rt * 16;
    int arow = tileRow + l15;
    if (arow >= nrows) arow = nrows - 1;        // clamp loads, guard stores
    const __bf16* ap = A + (size_t)arow * HDIM;
    v8f acc = {};
#pragma unroll
    for (int ks = 0; ks < 4; ++ks) {
      // A 16x32 fragment: lanes 0-15 hold K kk..kk+15,
      //                   lanes 16-31 hold K kk+16..kk+31
      v8bf alo = *(const v8bf*)(ap + ks * 32 + hi * 16);
      v8bf ahi = *(const v8bf*)(ap + ks * 32 + hi * 16 + 8);
      v16bf a = __builtin_shufflevector(alo, ahi,
                                        0,1,2,3,4,5,6,7,8,9,10,11,12,13,14,15);
      acc = __builtin_amdgcn_wmma_f32_16x16x32_bf16(false, a, false, bfrag[ks],
                                                    (short)0, acc, false, false);
    }
#pragma unroll
    for (int r = 0; r < 8; ++r) {
      int grow = tileRow + hi * 8 + r;          // C/D layout: VGPR r -> row
      if (grow < nrows) {
        float v = acc[r] + bv;
        if (do_relu) v = fmaxf(v, 0.0f);
        size_t o = (size_t)grow * HDIM + colBase + l15;
        C[o] = v;
        if (Cb) Cb[o] = (__bf16)v;
      }
    }
  }
}

// a_s[n] = xt[n].att_src ; a_d[n] = xt[n].att_dst  — one wave per node.
__global__ void __launch_bounds__(256)
a_sd_k(const float* __restrict__ xt, const float* __restrict__ asrc,
       const float* __restrict__ adst, float* __restrict__ a_s,
       float* __restrict__ a_d, int N) {
  int n = blockIdx.x * 8 + (threadIdx.x >> 5);
  int lane = threadIdx.x & 31;
  if (n >= N) return;
  const float4 xv = *(const float4*)(xt + (size_t)n * HDIM + lane * 4);
  const float4 s4 = *(const float4*)(asrc + lane * 4);
  const float4 d4 = *(const float4*)(adst + lane * 4);
  float ps = wave_sum(xv.x * s4.x + xv.y * s4.y + xv.z * s4.z + xv.w * s4.w);
  float pd = wave_sum(xv.x * d4.x + xv.y * d4.y + xv.z * d4.z + xv.w * d4.w);
  if (lane == 0) { a_s[n] = ps; a_d[n] = pd; }
}

// out[n][:] = bias; nmax[n] = -inf; nsum[n] = 0
__global__ void __launch_bounds__(256)
node_init_k(float* __restrict__ outb, float* __restrict__ nmax,
            float* __restrict__ nsum, const float* __restrict__ bias, int N) {
  long idx = (long)blockIdx.x * blockDim.x + threadIdx.x;
  long tot = (long)N * HDIM;
  if (idx >= tot) return;
  int d = (int)(idx & 127);
  outb[idx] = bias[d];
  if (d == 0) { int n = (int)(idx >> 7); nmax[n] = -INFINITY; nsum[n] = 0.0f; }
}

// alpha-logit = leaky_relu(a_s[src]+a_d[dst]+ea*c_l); record + segment max.
__global__ void __launch_bounds__(256)
edge_logit_max_k(const int* __restrict__ src, const int* __restrict__ dst,
                 const float* __restrict__ ea, const float* __restrict__ a_s,
                 const float* __restrict__ a_d, const float* __restrict__ scal,
                 int layer, float* __restrict__ elog, float* __restrict__ nmax,
                 int E, int EN) {
  int t = blockIdx.x * blockDim.x + threadIdx.x;
  if (t >= EN) return;
  int s, d; float e;
  if (t < E) { s = src[t]; d = dst[t]; e = ea[t]; }
  else       { s = t - E; d = s;       e = scal[0]; }   // self loop, mean fill
  float v = a_s[s] + a_d[d] + e * scal[1 + layer];
  v = (v > 0.0f) ? v : SLOPE * v;
  elog[t] = v;
  atomic_max_float(&nmax[d], v);
}

__global__ void __launch_bounds__(256)
edge_exp_sum_k(const int* __restrict__ dst, float* __restrict__ elog,
               const float* __restrict__ nmax, float* __restrict__ nsum,
               int E, int EN) {
  int t = blockIdx.x * blockDim.x + threadIdx.x;
  if (t >= EN) return;
  int d = (t < E) ? dst[t] : t - E;
  float ex = __expf(elog[t] - nmax[d]);
  elog[t] = ex;
  atomicAdd(&nsum[d], ex);
}

// out[dst] += alpha * xt[src]  — one wave per edge, 4 dims/lane.
__global__ void __launch_bounds__(256)
edge_scatter_k(const int* __restrict__ src, const int* __restrict__ dst,
               const float* __restrict__ elog, const float* __restrict__ nsum,
               const float* __restrict__ xt, float* __restrict__ outb,
               int E, int EN) {
  int w = blockIdx.x * 8 + (threadIdx.x >> 5);
  int lane = threadIdx.x & 31;
  if (w >= EN) return;
  int s, d;
  if (w < E) { s = src[w]; d = dst[w]; }
  else       { s = w - E; d = s; }
  float a = elog[w] / nsum[d];
  const float4 xv = *(const float4*)(xt + (size_t)s * HDIM + lane * 4);
  float* op = outb + (size_t)d * HDIM + lane * 4;
  atomicAdd(op + 0, xv.x * a);
  atomicAdd(op + 1, xv.y * a);
  atomicAdd(op + 2, xv.z * a);
  atomicAdd(op + 3, xv.w * a);
}

// h = out / max(||out||2, 1e-12); also emit bf16. One wave per node.
__global__ void __launch_bounds__(256)
node_normalize_k(const float* __restrict__ outb, float* __restrict__ h,
                 __bf16* __restrict__ hb, int N) {
  int n = blockIdx.x * 8 + (threadIdx.x >> 5);
  int lane = threadIdx.x & 31;
  if (n >= N) return;
  float4 v = *(const float4*)(outb + (size_t)n * HDIM + lane * 4);
  float ss = wave_sum(v.x * v.x + v.y * v.y + v.z * v.z + v.w * v.w);
  float inv = 1.0f / fmaxf(sqrtf(ss), 1e-12f);
  v.x *= inv; v.y *= inv; v.z *= inv; v.w *= inv;
  *(float4*)(h + (size_t)n * HDIM + lane * 4) = v;
  v4bf b; b[0] = (__bf16)v.x; b[1] = (__bf16)v.y; b[2] = (__bf16)v.z; b[3] = (__bf16)v.w;
  *(v4bf*)(hb + (size_t)n * HDIM + lane * 4) = b;
}

// pool[batch[n]] += h[n]  — one wave per node.
__global__ void __launch_bounds__(256)
pool_k(const float* __restrict__ h, const int* __restrict__ batch,
       float* __restrict__ pool, int N) {
  int n = blockIdx.x * 8 + (threadIdx.x >> 5);
  int lane = threadIdx.x & 31;
  if (n >= N) return;
  int g = batch[n];
  const float4 v = *(const float4*)(h + (size_t)n * HDIM + lane * 4);
  float* pp = pool + (size_t)g * HDIM + lane * 4;
  atomicAdd(pp + 0, v.x);
  atomicAdd(pp + 1, v.y);
  atomicAdd(pp + 2, v.z);
  atomicAdd(pp + 3, v.w);
}

// out[g] = pool[g].wp_W + wp_b  — one wave per graph.
__global__ void __launch_bounds__(256)
final_k(const float* __restrict__ pool, const float* __restrict__ wpW,
        const float* __restrict__ wpb, float* __restrict__ out, int G) {
  int g = blockIdx.x * 8 + (threadIdx.x >> 5);
  int lane = threadIdx.x & 31;
  if (g >= G) return;
  const float4 pv = *(const float4*)(pool + (size_t)g * HDIM + lane * 4);
  const float4 w4 = *(const float4*)(wpW + lane * 4);
  float s = wave_sum(pv.x * w4.x + pv.y * w4.y + pv.z * w4.z + pv.w * w4.w);
  if (lane == 0) out[g] = s + wpb[0];
}

// ---------------------------------------------------------------------------
extern "C" void kernel_launch(void* const* d_in, const int* in_sizes, int n_in,
                              void* d_out, int out_size, void* d_ws, size_t ws_size,
                              hipStream_t stream) {
  const int H  = HDIM;
  const int N  = in_sizes[0];
  const int E  = in_sizes[1] / 2;
  const int G  = out_size;
  const int L  = in_sizes[5]  / (H * H);
  const int LO = in_sizes[11] / (H * H);
  const int EN = E + N;

  const int*   x         = (const int*)d_in[0];
  const int*   src       = (const int*)d_in[1];
  const int*   dst       = src + E;
  const float* edge_attr = (const float*)d_in[2];
  const int*   batch     = (const int*)d_in[3];
  const float* emb       = (const float*)d_in[4];
  const float* gat_W     = (const float*)d_in[5];
  const float* att_src   = (const float*)d_in[6];
  const float* att_dst   = (const float*)d_in[7];
  const float* edge_W    = (const float*)d_in[8];
  const float* att_edge  = (const float*)d_in[9];
  const float* gat_b     = (const float*)d_in[10];
  const float* lin_W     = (const float*)d_in[11];
  const float* lin_b     = (const float*)d_in[12];
  const float* wp_W      = (const float*)d_in[13];
  const float* wp_b      = (const float*)d_in[14];
  float* out = (float*)d_out;
  (void)n_in; (void)ws_size;

  // Workspace carve (~190 MB for N=100k,E=400k,G=2048)
  char* base = (char*)d_ws;
  size_t off = 0;
  auto carve = [&](size_t bytes) -> char* {
    char* p = base + off;
    off = (off + bytes + 255) & ~(size_t)255;
    return p;
  };
  float*  h    = (float*) carve((size_t)N * H * 4);
  float*  xt   = (float*) carve((size_t)N * H * 4);
  float*  outb = (float*) carve((size_t)N * H * 4);
  __bf16* hb   = (__bf16*)carve((size_t)N * H * 2);
  __bf16* hb2  = (__bf16*)carve((size_t)N * H * 2);
  __bf16* WbT  = (__bf16*)carve((size_t)(L + LO) * H * H * 2);
  float*  a_s  = (float*) carve((size_t)N * 4);
  float*  a_d  = (float*) carve((size_t)N * 4);
  float*  elog = (float*) carve((size_t)EN * 4);
  float*  nmax = (float*) carve((size_t)N * 4);
  float*  nsum = (float*) carve((size_t)N * 4);
  float*  pool = (float*) carve((size_t)G * H * 4);
  float*  scal = (float*) carve(64);

  const int T = 256;

  // Prep
  zero_f<<<(G * H + T - 1) / T, T, 0, stream>>>(pool, G * H);
  zero_f<<<1, 32, 0, stream>>>(scal, 16);
  int rblocks = (E + T - 1) / T; if (rblocks > 1024) rblocks = 1024;
  reduce_sum_k<<<rblocks, T, 0, stream>>>(edge_attr, E, scal + 4);
  prep_scalars_k<<<1, (L + 1) * 32, 0, stream>>>(edge_W, att_edge, scal, L, E);
  convert_weights_k<<<((L + LO) * H * H + T - 1) / T, T, 0, stream>>>(gat_W, lin_W, WbT, L, LO);
  embed_k<<<((long)N * 32 + T - 1) / T, T, 0, stream>>>(x, emb, h, hb, N);

  const int gemmBlocks = (N + 63) / 64;

  // GAT layers
  for (int l = 0; l < L; ++l) {
    gemm128_bf16_wmma<<<gemmBlocks, T, 0, stream>>>(
        hb, WbT + (size_t)l * H * H, nullptr, xt, nullptr, N, 0);
    a_sd_k<<<(N + 7) / 8, T, 0, stream>>>(xt, att_src + (size_t)l * H,
                                          att_dst + (size_t)l * H, a_s, a_d, N);
    node_init_k<<<((long)N * H + T - 1) / T, T, 0, stream>>>(
        outb, nmax, nsum, gat_b + (size_t)l * H, N);
    edge_logit_max_k<<<(EN + T - 1) / T, T, 0, stream>>>(
        src, dst, edge_attr, a_s, a_d, scal, l, elog, nmax, E, EN);
    edge_exp_sum_k<<<(EN + T - 1) / T, T, 0, stream>>>(dst, elog, nmax, nsum, E, EN);
    edge_scatter_k<<<(EN + 7) / 8, T, 0, stream>>>(src, dst, elog, nsum, xt, outb, E, EN);
    node_normalize_k<<<(N + 7) / 8, T, 0, stream>>>(outb, h, hb, N);
  }

  // Dense layers (bias + ReLU fused in GEMM epilogue)
  const __bf16* cur = hb;
  __bf16* nxt = hb2;
  for (int i = 0; i < LO; ++i) {
    int last = (i == LO - 1);
    gemm128_bf16_wmma<<<gemmBlocks, T, 0, stream>>>(
        cur, WbT + (size_t)(L + i) * H * H, lin_b + (size_t)i * H,
        last ? h : xt, last ? (__bf16*)nullptr : nxt, N, 1);
    const __bf16* t = cur; cur = nxt; nxt = (__bf16*)t;
  }

  // Readout
  pool_k<<<(N + 7) / 8, T, 0, stream>>>(h, batch, pool, N);
  final_k<<<(G + 7) / 8, T, 0, stream>>>(pool, wp_W, wp_b, out, G);
}
